// Quantizer_12421045420160
// MI455X (gfx1250) — matrix operand order repaired and checked by reading, not compiled, not run on hardware
//
#include <hip/hip_runtime.h>

// ---------------- types ----------------
typedef __bf16 bf16_t;
typedef __attribute__((ext_vector_type(8)))  __bf16 v8bf;
typedef __attribute__((ext_vector_type(16))) __bf16 v16bf;
typedef __attribute__((ext_vector_type(8)))  float  v8f;
typedef __attribute__((ext_vector_type(4)))  unsigned int u32x4;
typedef __attribute__((ext_vector_type(8)))  int i32x8;
typedef __attribute__((ext_vector_type(4)))  int i32x4;

// ---------------- problem constants ----------------
#define NBATCH   32
#define EDIM     512
#define TDIM     2048
#define NE       8192
#define MTOT     (NBATCH * TDIM)                 // 65536 rows
#define ZQ_ELEMS ((size_t)NBATCH * EDIM * TDIM)  // 33554432
#define BETA     0.25f

// main-kernel tiling
#define BM 128             // rows per block
#define BN 64              // codes per TDM tile
#define NITER (NE / BN)    // 128

// LDS layout: 1040-byte row pitch (1024 data + 16 pad) -> conflict-free b128 frags
#define ROWP      1040
#define LDS_A_OFF 0
#define LDS_B0_OFF (BM * ROWP)                  // 133120
#define LDS_BUFSZ  (BN * ROWP)                  // 66560
#define LDS_B1_OFF (LDS_B0_OFF + LDS_BUFSZ)    // 199680
#define LDS_TOTAL  (LDS_B1_OFF + LDS_BUFSZ)    // 266240 (<320KB)

// workspace layout (bytes)
#define WS_EMBBF_OFF  0                        // 8192*512*2 = 8 MB
#define WS_ENORM_OFF  (8u * 1024u * 1024u)     // 8192*4
#define WS_IDX_OFF    (WS_ENORM_OFF + 32768u)  // 65536*4
#define WS_CNT_OFF    (WS_IDX_OFF + 262144u)   // 8192*4
#define WS_LOSS_OFF   (WS_CNT_OFF + 32768u)    // 1 float

// ---------------- helpers ----------------
__device__ __forceinline__ float wave_red_sum(float v) {
  #pragma unroll
  for (int off = 16; off > 0; off >>= 1) v += __shfl_down(v, off, 32);
  return v;
}

__device__ __forceinline__ v8f zero8() {
  v8f z = {0.f, 0.f, 0.f, 0.f, 0.f, 0.f, 0.f, 0.f};
  return z;
}

// A fragment: lane half g holds channels {k0+8g..+8} then {k0+16+8g..+8}
__device__ __forceinline__ v16bf ld_fragA(const char* sA, int rowoff,
                                          int cb, int half) {
  v8bf lo = *(const v8bf*)(sA + rowoff + ((cb + half) << 4));
  v8bf hi = *(const v8bf*)(sA + rowoff + ((cb + 2 + half) << 4));
  return __builtin_shufflevector(lo, hi, 0,1,2,3,4,5,6,7,8,9,10,11,12,13,14,15);
}

// B fragment: lane half g holds channels {k0+8g..+8} then {k0+16+8g..+8}
__device__ __forceinline__ v16bf ld_fragB(const char* sB, int rowoff,
                                          int cb, int half) {
  int c0 = cb + half;
  v8bf lo = *(const v8bf*)(sB + rowoff + (c0 << 4));
  v8bf hi = *(const v8bf*)(sB + rowoff + ((c0 + 2) << 4));
  return __builtin_shufflevector(lo, hi, 0,1,2,3,4,5,6,7,8,9,10,11,12,13,14,15);
}

// TDM: async 2D tile load  embbf[j0..j0+63][0..511] (bf16) -> LDS @ lds_off,
// with 4-DWORD pad every 256 DWORDs (row pitch 1040B). Tracked by TENSORcnt.
__device__ __forceinline__ void tdm_load_B(const bf16_t* gsrc, unsigned lds_off) {
  unsigned long long ga = (unsigned long long)gsrc;
  u32x4 g0;
  g0.x = 1u;                                   // count=1 (valid), user mode
  g0.y = lds_off;                              // LDS byte address
  g0.z = (unsigned)ga;                         // global_addr[31:0]
  g0.w = (unsigned)((ga >> 32) & 0x01FFFFFFull) | 0x80000000u;  // addr[56:32]|type=2
  i32x8 g1;
  g1[0] = (1 << 16)      // data_size = 2 bytes
        | (1 << 20)      // pad_enable
        | (7 << 22)      // pad_interval: every 256 DWORDs (=1 row of 1KB)
        | (3 << 25);     // pad_amount: 4 DWORDs (16B)
  g1[1] = (int)(512u << 16);    // tensor_dim0 = 512 (lo16)
  g1[2] = (int)(8192u << 16);   // tensor_dim0 hi=0 | tensor_dim1 = 8192 (lo16)
  g1[3] = (int)(512u << 16);    // tensor_dim1 hi=0 | tile_dim0 = 512
  g1[4] = BN;                   // tile_dim1 = 64, tile_dim2 = 0
  g1[5] = 512;                  // tensor_dim0_stride lo32 = 512 elements
  g1[6] = 0;                    // stride0 hi | stride1 lo
  g1[7] = 0;                    // stride1 hi
  i32x4 gz = {0, 0, 0, 0};
#if defined(__clang_major__) && (__clang_major__ >= 23)
  i32x8 gz8 = {0, 0, 0, 0, 0, 0, 0, 0};
  __builtin_amdgcn_tensor_load_to_lds(g0, g1, gz, gz, gz8, 0);
#else
  __builtin_amdgcn_tensor_load_to_lds(g0, g1, gz, gz, 0);
#endif
}

// ---------------- kernel 1: codebook prep ----------------
__global__ __launch_bounds__(256) void prep_emb_kernel(
    const float* __restrict__ emb, bf16_t* __restrict__ embbf,
    float* __restrict__ enorm, unsigned* __restrict__ counts,
    float* __restrict__ loss_acc) {
  __shared__ float part[8];
  const int j = blockIdx.x;
  const int tid = threadIdx.x;
  float s = 0.f;
  #pragma unroll
  for (int c = tid; c < EDIM; c += 256) {
    float v = emb[(size_t)j * EDIM + c];
    embbf[(size_t)j * EDIM + c] = (bf16_t)v;
    s += v * v;
  }
  s = wave_red_sum(s);
  if ((tid & 31) == 0) part[tid >> 5] = s;
  __syncthreads();
  if (tid == 0) {
    float tot = 0.f;
    #pragma unroll
    for (int w = 0; w < 8; ++w) tot += part[w];
    enorm[j] = tot;
    counts[j] = 0u;
    if (j == 0) *loss_acc = 0.f;
  }
}

// ---------------- kernel 2: WMMA score GEMM + running argmin ----------------
__global__ __launch_bounds__(256) void vq_argmin_kernel(
    const float* __restrict__ z, const bf16_t* __restrict__ embbf,
    const float* __restrict__ enorm, int* __restrict__ idx_out) {
  extern __shared__ char smem[];
  char* sA = smem + LDS_A_OFF;

  const int tid  = threadIdx.x;
  const int lane = tid & 31;
  const int wid  = tid >> 5;
  const int wm   = wid >> 1;       // 0..3  (M wave): msubs {2wm, 2wm+1}
  const int wn   = wid & 1;        // 0..1  (N wave): nsubs {2wn, 2wn+1}
  const int l16  = lane & 15;
  const int half = lane >> 4;

  const int m0 = blockIdx.x * BM;            // rows m0..m0+127 share batch index
  const int nb = m0 / TDIM;
  const int t0 = m0 % TDIM;
  const float* zbase = z + (size_t)nb * EDIM * TDIM + t0;

  // ---- stage A once: z[nb, c, t0+r] -> bf16, pitch-1040 LDS ----
  for (int e = tid; e < BM * EDIM; e += 256) {
    int r = e & (BM - 1);
    int c = e >> 7;                            // coalesced over r (t contiguous)
    float v = zbase[(size_t)c * TDIM + r];
    *(bf16_t*)(sA + r * ROWP + (c << 1)) = (bf16_t)v;
  }

  // per-lane running argmin
  float bestd[2][8];
  int   bestj[2][8];
  #pragma unroll
  for (int ms = 0; ms < 2; ++ms)
    #pragma unroll
    for (int v = 0; v < 8; ++v) { bestd[ms][v] = 3.4e38f; bestj[ms][v] = 0; }

  const int rowA0 = ((wm * 2 + 0) * 16 + l16) * ROWP;
  const int rowA1 = ((wm * 2 + 1) * 16 + l16) * ROWP;
  const int rowB0 = ((wn * 2 + 0) * 16 + l16) * ROWP;
  const int rowB1 = ((wn * 2 + 1) * 16 + l16) * ROWP;

  // ---- prologue: TDM-load tile 0 into buffer 0 ----
  if (wid == 0) {
    tdm_load_B(embbf, LDS_B0_OFF);
    __builtin_amdgcn_s_wait_tensorcnt(0);
  }
  __syncthreads();

  for (int it = 0; it < NITER; ++it) {
    const int j0 = it * BN;
    const char* sB = smem + ((it & 1) ? LDS_B1_OFF : LDS_B0_OFF);

    // kick off DMA of the next tile into the other buffer (overlaps compute)
    if (wid == 0 && (it + 1) < NITER)
      tdm_load_B(embbf + (size_t)(j0 + BN) * EDIM,
                 ((it + 1) & 1) ? LDS_B1_OFF : LDS_B0_OFF);

    v8f acc00 = zero8(), acc01 = zero8(), acc10 = zero8(), acc11 = zero8();

    // software-pipelined K loop: fragments for kc+1 in flight during wmma(kc)
    v16bf a0c = ld_fragA(sA, rowA0, 0, half);
    v16bf a1c = ld_fragA(sA, rowA1, 0, half);
    v16bf b0c = ld_fragB(sB, rowB0, 0, half);
    v16bf b1c = ld_fragB(sB, rowB1, 0, half);
    #pragma unroll
    for (int kc = 0; kc < EDIM / 32; ++kc) {
      v16bf a0n = a0c, a1n = a1c, b0n = b0c, b1n = b1c;
      if (kc < EDIM / 32 - 1) {
        const int cbn = (kc + 1) * 4;
        a0n = ld_fragA(sA, rowA0, cbn, half);
        a1n = ld_fragA(sA, rowA1, cbn, half);
        b0n = ld_fragB(sB, rowB0, cbn, half);
        b1n = ld_fragB(sB, rowB1, cbn, half);
      }
      acc00 = __builtin_amdgcn_wmma_f32_16x16x32_bf16(false, a0c, false, b0c,
                                                      (short)0, acc00, false, false);
      acc01 = __builtin_amdgcn_wmma_f32_16x16x32_bf16(false, a0c, false, b1c,
                                                      (short)0, acc01, false, false);
      acc10 = __builtin_amdgcn_wmma_f32_16x16x32_bf16(false, a1c, false, b0c,
                                                      (short)0, acc10, false, false);
      acc11 = __builtin_amdgcn_wmma_f32_16x16x32_bf16(false, a1c, false, b1c,
                                                      (short)0, acc11, false, false);
      a0c = a0n; a1c = a1n; b0c = b0n; b1c = b1n;
    }

    // ---- score (d = ||e||^2 - 2*dot; ||z||^2 dropped) + running argmin ----
    const int jc0 = j0 + (wn * 2 + 0) * 16 + l16;
    const int jc1 = jc0 + 16;
    const float en0 = enorm[jc0];
    const float en1 = enorm[jc1];
    #pragma unroll
    for (int v = 0; v < 8; ++v) {
      float d;
      d = fmaf(-2.f, acc00[v], en0);
      if (d < bestd[0][v]) { bestd[0][v] = d; bestj[0][v] = jc0; }
      d = fmaf(-2.f, acc01[v], en1);
      if (d < bestd[0][v]) { bestd[0][v] = d; bestj[0][v] = jc1; }
      d = fmaf(-2.f, acc10[v], en0);
      if (d < bestd[1][v]) { bestd[1][v] = d; bestj[1][v] = jc0; }
      d = fmaf(-2.f, acc11[v], en1);
      if (d < bestd[1][v]) { bestd[1][v] = d; bestj[1][v] = jc1; }
    }

    // drain this iteration's DMA, then rendezvous (buffer swap safety)
    if (wid == 0) __builtin_amdgcn_s_wait_tensorcnt(0);
    __syncthreads();
  }

  // ---- butterfly argmin across the 16 lanes sharing a row ----
  float fd[2][8]; int fj[2][8];
  #pragma unroll
  for (int ms = 0; ms < 2; ++ms)
    #pragma unroll
    for (int v = 0; v < 8; ++v) {
      float d = bestd[ms][v]; int j = bestj[ms][v];
      #pragma unroll
      for (int mk = 1; mk <= 8; mk <<= 1) {
        float d2 = __shfl_xor(d, mk, 32);
        int   j2 = __shfl_xor(j, mk, 32);
        if (d2 < d || (d2 == d && j2 < j)) { d = d2; j = j2; }
      }
      fd[ms][v] = d; fj[ms][v] = j;
    }

  // ---- combine across the 2 N-waves via (reused) LDS, write indices ----
  float* scr_d = (float*)sA;            // [128][2]
  int*   scr_j = (int*)(sA + 1024);     // [128][2]
  if (l16 == 0) {
    #pragma unroll
    for (int ms = 0; ms < 2; ++ms)
      #pragma unroll
      for (int v = 0; v < 8; ++v) {
        int row = (wm * 2 + ms) * 16 + v + 8 * half;
        scr_d[row * 2 + wn] = fd[ms][v];
        scr_j[row * 2 + wn] = fj[ms][v];
      }
  }
  __syncthreads();
  if (tid < BM) {
    float bd = scr_d[tid * 2]; int bj = scr_j[tid * 2];
    float d2 = scr_d[tid * 2 + 1]; int j2 = scr_j[tid * 2 + 1];
    if (d2 < bd || (d2 == bd && j2 < bj)) { bd = d2; bj = j2; }
    idx_out[m0 + tid] = bj;
  }
}

// ---------------- kernel 3: gather z_q (N,C,T) + loss partial sums ----------------
__global__ __launch_bounds__(256) void gather_kernel(
    const float* __restrict__ z, const float* __restrict__ emb,
    const int* __restrict__ idx, float* __restrict__ out,
    float* __restrict__ loss_acc) {
  __shared__ float part[8];
  const int tid = threadIdx.x;
  size_t o = (size_t)blockIdx.x * 256 + tid;
  int t = (int)(o & (TDIM - 1));
  int c = (int)((o >> 11) & (EDIM - 1));
  int nb = (int)(o >> 20);
  int m = (nb << 11) + t;
  int id = idx[m];
  float e  = emb[(size_t)id * EDIM + c];
  float zz = z[o];
  out[o] = zz + (e - zz);              // straight-through, same arithmetic as ref
  float diff = e - zz;
  float s = wave_red_sum(diff * diff);
  if ((tid & 31) == 0) part[tid >> 5] = s;
  __syncthreads();
  if (tid == 0) {
    float tot = 0.f;
    #pragma unroll
    for (int w = 0; w < 8; ++w) tot += part[w];
    atomicAdd(loss_acc, tot);
  }
}

// ---------------- kernel 4: code-usage histogram ----------------
__global__ __launch_bounds__(256) void hist_kernel(
    const int* __restrict__ idx, unsigned* __restrict__ counts) {
  int m = blockIdx.x * 256 + threadIdx.x;
  if (m < MTOT) atomicAdd(&counts[idx[m]], 1u);
}

// ---------------- kernel 5: loss + perplexity ----------------
__global__ __launch_bounds__(256) void finalize_kernel(
    const unsigned* __restrict__ counts, const float* __restrict__ loss_acc,
    float* __restrict__ out_scalars) {
  __shared__ float red[256];
  const int tid = threadIdx.x;
  float h = 0.f;
  for (int j = tid; j < NE; j += 256) {
    float em = (float)counts[j] * (1.0f / (float)MTOT);
    h += em * logf(em + 1e-10f);
  }
  red[tid] = h;
  __syncthreads();
  for (int s = 128; s > 0; s >>= 1) {
    if (tid < s) red[tid] += red[tid + s];
    __syncthreads();
  }
  if (tid == 0) {
    float loss = loss_acc[0] * ((1.0f + BETA) / (float)ZQ_ELEMS);
    out_scalars[0] = loss;
    out_scalars[1] = expf(-red[0]);
  }
}

// ---------------- launcher ----------------
extern "C" void kernel_launch(void* const* d_in, const int* in_sizes, int n_in,
                              void* d_out, int out_size, void* d_ws, size_t ws_size,
                              hipStream_t stream) {
  const float* z   = (const float*)d_in[0];   // (32, 512, 2048) f32
  const float* emb = (const float*)d_in[1];   // (8192, 512) f32
  float* out = (float*)d_out;                 // z_q (33554432) + loss + perplexity

  char* ws = (char*)d_ws;                     // ~8.8 MB used
  bf16_t*   embbf    = (bf16_t*)(ws + WS_EMBBF_OFF);
  float*    enorm    = (float*)(ws + WS_ENORM_OFF);
  int*      idxb     = (int*)(ws + WS_IDX_OFF);
  unsigned* counts   = (unsigned*)(ws + WS_CNT_OFF);
  float*    loss_acc = (float*)(ws + WS_LOSS_OFF);

  prep_emb_kernel<<<NE, 256, 0, stream>>>(emb, embbf, enorm, counts, loss_acc);

  // 266240 B dynamic LDS: legal on CDNA5 (320 KB per workgroup)
  vq_argmin_kernel<<<MTOT / BM, 256, LDS_TOTAL, stream>>>(z, embbf, enorm, idxb);

  gather_kernel<<<(unsigned)(ZQ_ELEMS / 256), 256, 0, stream>>>(
      z, emb, idxb, out, loss_acc);

  hist_kernel<<<MTOT / 256, 256, 0, stream>>>(idxb, counts);

  finalize_kernel<<<1, 256, 0, stream>>>(counts, loss_acc, out + ZQ_ELEMS);
}
